// MultiHeadAttention_76905684402869
// MI455X (gfx1250) — matrix operand order repaired and checked by reading, compiled, not run on hardware
//
#include <hip/hip_runtime.h>
#include <hip/hip_bf16.h>

// ---------------- problem constants ----------------
#define N_HEAD   16
#define D_K      64
#define D_V      64
#define D_MODEL  1024
#define BATCH    4
#define SEQ      2048
#define M_TOT    (BATCH * SEQ)         // 8192 token rows
#define LN_EPS   1e-6f

// ---------------- vector types ----------------
typedef __attribute__((ext_vector_type(16))) __bf16 v16bf;
typedef __attribute__((ext_vector_type(8)))  __bf16 v8bf;
typedef __attribute__((ext_vector_type(4)))  __bf16 v4bf;
typedef __attribute__((ext_vector_type(8)))  float  v8f;
typedef __attribute__((ext_vector_type(4)))  int    v4i;

// ---- gfx1250 async global->LDS path (guarded; falls back to sync staging) ----
#if __has_builtin(__builtin_amdgcn_global_load_async_to_lds_b128) && \
    __has_builtin(__builtin_amdgcn_s_wait_asynccnt)
#define HAVE_ASYNC_LDS 1
#define WAIT_ASYNC(n) __builtin_amdgcn_s_wait_asynccnt(n)
#else
#define HAVE_ASYNC_LDS 0
#define WAIT_ASYNC(n)
#endif

static __device__ __forceinline__ __bf16 f2bf(float f) { return (__bf16)f; }

static __device__ __forceinline__ v8f wmma_bf16(v16bf a, v16bf b, v8f c) {
  // D = A(16x32 bf16) x B(32x16 bf16) + C(16x16 f32)
  return __builtin_amdgcn_wmma_f32_16x16x32_bf16(
      false, a, false, b, (short)0, c, false, false);
}

// ---- A-fragment (16x32, ISA 7.12.2 16-bit A layout) from bf16 row-major ----
static __device__ __forceinline__ v16bf load_a_bf16(const __bf16* A, int ld,
                                                    int row0, int k0, int lane) {
  const int m  = row0 + (lane & 15);
  const int hi = (lane >> 4) & 1;
  const __bf16* p = A + (size_t)m * ld + k0 + hi * 8;
  v8bf lo = *(const v8bf*)(p);
  v8bf hh = *(const v8bf*)(p + 16);
  return __builtin_shufflevector(lo, hh, 0,1,2,3,4,5,6,7,8,9,10,11,12,13,14,15);
}

// ---- A-fragment from f32 row-major (P matrix read back from d_out) ----
static __device__ __forceinline__ v16bf load_a_f32(const float* A, int ld,
                                                   int row0, int k0, int lane) {
  const int m  = row0 + (lane & 15);
  const int hi = (lane >> 4) & 1;
  const float* p = A + (size_t)m * ld + k0 + hi * 8;
  float4 f0 = *(const float4*)(p + 0);
  float4 f1 = *(const float4*)(p + 4);
  float4 f2 = *(const float4*)(p + 16);
  float4 f3 = *(const float4*)(p + 20);
  v16bf a;
  a[0]=f2bf(f0.x);  a[1]=f2bf(f0.y);  a[2]=f2bf(f0.z);  a[3]=f2bf(f0.w);
  a[4]=f2bf(f1.x);  a[5]=f2bf(f1.y);  a[6]=f2bf(f1.z);  a[7]=f2bf(f1.w);
  a[8]=f2bf(f2.x);  a[9]=f2bf(f2.y);  a[10]=f2bf(f2.z); a[11]=f2bf(f2.w);
  a[12]=f2bf(f3.x); a[13]=f2bf(f3.y); a[14]=f2bf(f3.z); a[15]=f2bf(f3.w);
  return a;
}

// ---- B-fragment (32x16) from K-contiguous bf16: elem(kd,n)=T[(n0+n)*ld+k0+kd] ----
static __device__ __forceinline__ v16bf load_bT(const __bf16* T, int ld,
                                                int n0, int k0, int lane) {
  const int n  = n0 + (lane & 15);
  const int hi = (lane >> 4) & 1;
  const __bf16* p = T + (size_t)n * ld + k0 + hi * 8;
  v8bf lo = *(const v8bf*)(p);
  v8bf hh = *(const v8bf*)(p + 16);
  return __builtin_shufflevector(lo, hh, 0,1,2,3,4,5,6,7,8,9,10,11,12,13,14,15);
}

// =====================================================================
// Pre-pass A: f32 -> bf16 bulk convert
// =====================================================================
__global__ void cvt_bf16_kernel(const float* __restrict__ src,
                                __bf16* __restrict__ dst, int n) {
  const int i = (blockIdx.x * blockDim.x + threadIdx.x) * 4;
  if (i >= n) return;
  float4 f = *(const float4*)(src + i);
  v4bf o;
  o[0] = f2bf(f.x); o[1] = f2bf(f.y); o[2] = f2bf(f.z); o[3] = f2bf(f.w);
  *(v4bf*)(dst + i) = o;
}

// =====================================================================
// Pre-pass B: W f32 [K][N] -> Wt bf16 [N][K] (LDS tile transpose)
// =====================================================================
__global__ void wT_kernel(const float* __restrict__ W, __bf16* __restrict__ Wt) {
  __shared__ __bf16 tile[32][33];
  const int bn = blockIdx.x * 32;
  const int bk = blockIdx.y * 32;
  const int tx = threadIdx.x & 31;
  const int t0 = threadIdx.x >> 5;
  for (int ty = t0; ty < 32; ty += 8)
    tile[ty][tx] = f2bf(W[(size_t)(bk + ty) * D_MODEL + bn + tx]);
  __syncthreads();
  for (int ty = t0; ty < 32; ty += 8)
    Wt[(size_t)(bn + ty) * D_MODEL + bk + tx] = tile[tx][ty];
}

// =====================================================================
// Kernel 1: projection GEMM  C = xb @ WtT,  block tile 32(M) x 128(N),
// 4 waves; shared A panel double-buffered in LDS via async loads.
//   vmode==0 : write bf16 [b,h,s,d]   (Q, K)
//   vmode==1 : write bf16 [b,h,d,s]   (V)
// grid (N/128, M/32) = (8, 256), block 128
// =====================================================================
__global__ void proj_kernel(const __bf16* __restrict__ xb,
                            const __bf16* __restrict__ wT,   // [N][K] bf16
                            __bf16* __restrict__ out, int vmode) {
  __shared__ __bf16 atile[2][32][32];
  const int tid  = threadIdx.x;
  const int wave = tid >> 5;
  const int lane = tid & 31;
  const int m0 = blockIdx.y * 32;
  const int nw = blockIdx.x * 128 + wave * 32;

  // staging map: thread -> (row, 16B segment)
  const int sr   = tid >> 2;
  const int sseg = tid & 3;
  const __bf16* sgp = xb + (size_t)(m0 + sr) * D_MODEL + sseg * 8;

  auto stage = [&](int buf, int kk) {
#if HAVE_ASYNC_LDS
    __builtin_amdgcn_global_load_async_to_lds_b128(
        (__attribute__((address_space(1))) v4i*)(sgp + kk),
        (__attribute__((address_space(3))) v4i*)&atile[buf][sr][sseg * 8],
        0, 0);
#else
    *(v8bf*)&atile[buf][sr][sseg * 8] = *(const v8bf*)(sgp + kk);
#endif
  };
  auto lds_a = [&](int buf, int row0) -> v16bf {
    const int m  = row0 + (lane & 15);
    const int hi = (lane >> 4) & 1;
    const __bf16* p = &atile[buf][m][hi * 8];
    v8bf lo = *(const v8bf*)(p);
    v8bf hh = *(const v8bf*)(p + 16);
    return __builtin_shufflevector(lo, hh,
                                   0,1,2,3,4,5,6,7,8,9,10,11,12,13,14,15);
  };

  v8f c00 = {}, c01 = {}, c10 = {}, c11 = {};
  int cur = 0;
  stage(0, 0);
  for (int kk = 0; kk < D_MODEL - 32; kk += 32) {
    stage(cur ^ 1, kk + 32);
    WAIT_ASYNC(1);                 // current buffer complete (in-order asyncs)
    __syncthreads();
    v16bf a0 = lds_a(cur, 0);
    v16bf a1 = lds_a(cur, 16);
    v16bf b0 = load_bT(wT, D_MODEL, nw, kk, lane);
    v16bf b1 = load_bT(wT, D_MODEL, nw + 16, kk, lane);
    c00 = wmma_bf16(a0, b0, c00);
    c01 = wmma_bf16(a0, b1, c01);
    c10 = wmma_bf16(a1, b0, c10);
    c11 = wmma_bf16(a1, b1, c11);
    __syncthreads();
    cur ^= 1;
  }
  {
    const int kk = D_MODEL - 32;
    WAIT_ASYNC(0);
    __syncthreads();
    v16bf a0 = lds_a(cur, 0);
    v16bf a1 = lds_a(cur, 16);
    v16bf b0 = load_bT(wT, D_MODEL, nw, kk, lane);
    v16bf b1 = load_bT(wT, D_MODEL, nw + 16, kk, lane);
    c00 = wmma_bf16(a0, b0, c00);
    c01 = wmma_bf16(a0, b1, c01);
    c10 = wmma_bf16(a1, b0, c10);
    c11 = wmma_bf16(a1, b1, c11);
  }

  const int colq = lane & 15;
  const int hi   = (lane >> 4) & 1;
  auto store_one = [&](v8f cc, int rbase, int cbase) {
    const int col = cbase + colq;
    const int h   = col >> 6;
    const int d   = col & 63;
#pragma unroll
    for (int r = 0; r < 8; ++r) {
      const int mrow = rbase + r + hi * 8;
      const int b_   = mrow >> 11;
      const int s_   = mrow & (SEQ - 1);
      const size_t head = (size_t)(b_ * N_HEAD + h);
      if (vmode)
        out[(head * D_V + d) * SEQ + s_] = f2bf(cc[r]);
      else
        out[(head * SEQ + s_) * D_K + d] = f2bf(cc[r]);
    }
  };
  store_one(c00, m0,      nw);
  store_one(c01, m0,      nw + 16);
  store_one(c10, m0 + 16, nw);
  store_one(c11, m0 + 16, nw + 16);
}

// =====================================================================
// Kernel 2: attention for one (b, h, 16-query strip).
// block = 256 threads (8 waves), grid (SEQ/16, N_HEAD, BATCH)
// =====================================================================
__global__ void attn_kernel(const __bf16* __restrict__ q,   // [b,h,s,d]
                            const __bf16* __restrict__ k,   // [b,h,s,d]
                            const __bf16* __restrict__ vt,  // [b,h,d,s]
                            float* __restrict__ attn,       // [b,h,S,S]
                            __bf16* __restrict__ ctx) {     // [M, 1024] bf16
  const int q0 = blockIdx.x * 16;
  const int h  = blockIdx.y;
  const int b  = blockIdx.z;
  const int tid  = threadIdx.x;
  const int wave = tid >> 5;
  const int lane = tid & 31;

  const __bf16* Qh = q  + ((size_t)(b * N_HEAD + h)) * SEQ * D_K;
  const __bf16* Kh = k  + ((size_t)(b * N_HEAD + h)) * SEQ * D_K;
  const __bf16* Vt = vt + ((size_t)(b * N_HEAD + h)) * D_V * SEQ;
  float* attn_h = attn + (((size_t)(b * N_HEAD + h)) * SEQ + q0) * SEQ;

  __shared__ float red[16][17];
  __shared__ float rowm[16];
  __shared__ float rowinv[16];
  __shared__ float ctx_acc[16][64];

  for (int i = tid; i < 16 * 64; i += 256) ((float*)ctx_acc)[i] = 0.0f;

  // ---------------- phase 1: raw scaled scores via WMMA ----------------
  const v16bf qa0 = load_a_bf16(Qh, D_K, q0, 0, lane);
  const v16bf qa1 = load_a_bf16(Qh, D_K, q0, 32, lane);
  const int colq = lane & 15;
  const int hi   = (lane >> 4) & 1;
  for (int kt = wave; kt < SEQ / 16; kt += 8) {
    const int key0 = kt * 16;
    v16bf b0 = load_bT(Kh, D_K, key0, 0, lane);   // B(kd,n)=K[key0+n][kd]
    v16bf b1 = load_bT(Kh, D_K, key0, 32, lane);
    v8f c = {};
    c = wmma_bf16(qa0, b0, c);
    c = wmma_bf16(qa1, b1, c);
#pragma unroll
    for (int r = 0; r < 8; ++r)
      attn_h[(size_t)(r + hi * 8) * SEQ + key0 + colq] = c[r] * 0.125f; // 1/sqrt(64)
  }
  __threadfence();
  __syncthreads();

  // ---------------- phase 1.5: row max and exp-sum (strip L2-resident) --------
  const int row = tid >> 4;
  const int sub = tid & 15;
  float* arow = attn_h + (size_t)row * SEQ;
  float mx = -3.4e38f;
  for (int j = sub; j < SEQ; j += 16) mx = fmaxf(mx, arow[j]);
  red[row][sub] = mx;
  __syncthreads();
  if (sub == 0) {
    float m = red[row][0];
#pragma unroll
    for (int i = 1; i < 16; ++i) m = fmaxf(m, red[row][i]);
    rowm[row] = m;
  }
  __syncthreads();
  const float m = rowm[row];
  float sum = 0.0f;
  for (int j = sub; j < SEQ; j += 16) sum += __expf(arow[j] - m);
  red[row][sub] = sum;
  __syncthreads();
  if (sub == 0) {
    float s = red[row][0];
#pragma unroll
    for (int i = 1; i < 16; ++i) s += red[row][i];
    rowinv[row] = 1.0f / s;
  }
  __syncthreads();

  // ---------------- phase 2a: normalize in place ----------------
  const float inv = rowinv[row];
  for (int j = sub; j < SEQ; j += 16) arow[j] = __expf(arow[j] - m) * inv;
  __threadfence();
  __syncthreads();

  // ---------------- phase 2b: ctx = P @ V via WMMA, K split over 8 waves ------
  v8f c0 = {}, c1 = {}, c2 = {}, c3 = {};
  for (int kk = wave * 32; kk < SEQ; kk += 8 * 32) {
    v16bf pa  = load_a_f32(attn_h, SEQ, 0, kk, lane);
    v16bf vb0 = load_bT(Vt, SEQ, 0,  kk, lane);   // B(kd,n)=Vt[n][kk+kd]
    v16bf vb1 = load_bT(Vt, SEQ, 16, kk, lane);
    v16bf vb2 = load_bT(Vt, SEQ, 32, kk, lane);
    v16bf vb3 = load_bT(Vt, SEQ, 48, kk, lane);
    c0 = wmma_bf16(pa, vb0, c0);
    c1 = wmma_bf16(pa, vb1, c1);
    c2 = wmma_bf16(pa, vb2, c2);
    c3 = wmma_bf16(pa, vb3, c3);
  }
#pragma unroll
  for (int r = 0; r < 8; ++r) {
    const int rr = r + hi * 8;
    atomicAdd(&ctx_acc[rr][0  + colq], c0[r]);
    atomicAdd(&ctx_acc[rr][16 + colq], c1[r]);
    atomicAdd(&ctx_acc[rr][32 + colq], c2[r]);
    atomicAdd(&ctx_acc[rr][48 + colq], c3[r]);
  }
  __syncthreads();

  // ---------------- phase 3: ctx tile -> bf16 [M, 1024] ----------------
  for (int i = tid; i < 16 * 64; i += 256) {
    const int r2 = i >> 6;
    const int d2 = i & 63;
    ctx[((size_t)(b * SEQ + q0 + r2)) * D_MODEL + h * D_V + d2] =
        f2bf(ctx_acc[r2][d2]);
  }
}

// =====================================================================
// Kernel 3: output projection + residual, 32x32 register-blocked.
// grid (N/32, M/32) = (32, 256), block = 32
// =====================================================================
__global__ void oproj_kernel(const __bf16* __restrict__ ctx,
                             const __bf16* __restrict__ woT,  // [N][K] bf16
                             const float* __restrict__ x,
                             float* __restrict__ pre) {
  const int lane = threadIdx.x & 31;
  const int n0 = blockIdx.x * 32;
  const int m0 = blockIdx.y * 32;
  v8f c00 = {}, c01 = {}, c10 = {}, c11 = {};
#pragma unroll 4
  for (int kk = 0; kk < D_MODEL; kk += 32) {
    v16bf a0 = load_a_bf16(ctx, D_MODEL, m0,      kk, lane);
    v16bf a1 = load_a_bf16(ctx, D_MODEL, m0 + 16, kk, lane);
    v16bf b0 = load_bT(woT, D_MODEL, n0,      kk, lane);
    v16bf b1 = load_bT(woT, D_MODEL, n0 + 16, kk, lane);
    c00 = wmma_bf16(a0, b0, c00);
    c01 = wmma_bf16(a0, b1, c01);
    c10 = wmma_bf16(a1, b0, c10);
    c11 = wmma_bf16(a1, b1, c11);
  }
  const int colq = lane & 15;
  const int hi   = (lane >> 4) & 1;
  auto store_one = [&](v8f cc, int rbase, int cbase) {
    const int col = cbase + colq;
#pragma unroll
    for (int r = 0; r < 8; ++r) {
      const int mrow = rbase + r + hi * 8;
      pre[(size_t)mrow * D_MODEL + col] = cc[r] + x[(size_t)mrow * D_MODEL + col];
    }
  };
  store_one(c00, m0,      n0);
  store_one(c01, m0,      n0 + 16);
  store_one(c10, m0 + 16, n0);
  store_one(c11, m0 + 16, n0 + 16);
}

// =====================================================================
// Kernel 4: LayerNorm over D_MODEL, one block per row
// =====================================================================
__global__ void ln_kernel(const float* __restrict__ pre,
                          const float* __restrict__ gamma,
                          const float* __restrict__ beta,
                          float* __restrict__ out) {
  const int rowi = blockIdx.x;
  const int tid  = threadIdx.x;
  const float* r = pre + (size_t)rowi * D_MODEL;
  float s = 0.0f, ss = 0.0f;
  for (int j = tid; j < D_MODEL; j += 256) {
    const float v0 = r[j];
    s += v0; ss += v0 * v0;
  }
  __shared__ float s1[256], s2[256];
  s1[tid] = s; s2[tid] = ss;
  __syncthreads();
  for (int off = 128; off > 0; off >>= 1) {
    if (tid < off) { s1[tid] += s1[tid + off]; s2[tid] += s2[tid + off]; }
    __syncthreads();
  }
  const float mean = s1[0] * (1.0f / D_MODEL);
  const float var  = s2[0] * (1.0f / D_MODEL) - mean * mean;
  const float rstd = rsqrtf(var + LN_EPS);
  for (int j = tid; j < D_MODEL; j += 256)
    out[(size_t)rowi * D_MODEL + j] = (r[j] - mean) * rstd * gamma[j] + beta[j];
}

// =====================================================================
// host-side launch
// =====================================================================
extern "C" void kernel_launch(void* const* d_in, const int* in_sizes, int n_in,
                              void* d_out, int out_size, void* d_ws, size_t ws_size,
                              hipStream_t stream) {
  (void)in_sizes; (void)n_in; (void)out_size; (void)ws_size;
  const float* x     = (const float*)d_in[0];
  const float* w_q   = (const float*)d_in[1];
  const float* w_k   = (const float*)d_in[2];
  const float* w_v   = (const float*)d_in[3];
  const float* w_o   = (const float*)d_in[4];
  const float* gamma = (const float*)d_in[5];
  const float* beta  = (const float*)d_in[6];

  float* out_ptr  = (float*)d_out;                                   // [B,S,D_MODEL]
  float* attn_ptr = out_ptr + (size_t)M_TOT * D_MODEL;               // [B,H,S,S]

  const size_t ME = (size_t)M_TOT * D_MODEL;                         // 8388608
  const size_t WE = (size_t)D_MODEL * D_MODEL;                       // 1048576
  __bf16* xb    = (__bf16*)d_ws;        // [M][1024]
  __bf16* wqT   = xb   + ME;            // [N][K]
  __bf16* wkT   = wqT  + WE;
  __bf16* wvT   = wkT  + WE;
  __bf16* woT   = wvT  + WE;
  __bf16* qws   = woT  + WE;            // [b,h,s,d]
  __bf16* kws   = qws  + ME;
  __bf16* vws   = kws  + ME;            // [b,h,d,s]
  __bf16* ctxws = vws  + ME;            // [M][1024]
  float*  pre   = (float*)(ctxws + ME);

  // ---- pre-pass ----
  cvt_bf16_kernel<<<(int)(ME / 4 / 256), 256, 0, stream>>>(x, xb, (int)ME);
  dim3 gT(D_MODEL / 32, D_MODEL / 32);
  wT_kernel<<<gT, 256, 0, stream>>>(w_q, wqT);
  wT_kernel<<<gT, 256, 0, stream>>>(w_k, wkT);
  wT_kernel<<<gT, 256, 0, stream>>>(w_v, wvT);
  wT_kernel<<<gT, 256, 0, stream>>>(w_o, woT);

  // ---- projections (4-wave blocks, async LDS-staged A panel) ----
  dim3 gproj(D_MODEL / 128, M_TOT / 32);  // (8, 256)
  proj_kernel<<<gproj, 128, 0, stream>>>(xb, wqT, qws, 0);
  proj_kernel<<<gproj, 128, 0, stream>>>(xb, wkT, kws, 0);
  proj_kernel<<<gproj, 128, 0, stream>>>(xb, wvT, vws, 1);

  // ---- attention ----
  dim3 gattn(SEQ / 16, N_HEAD, BATCH);    // (128, 16, 4)
  attn_kernel<<<gattn, 256, 0, stream>>>(qws, kws, vws, attn_ptr, ctxws);

  // ---- output projection + residual, then LayerNorm ----
  dim3 goproj(D_MODEL / 32, M_TOT / 32);  // (32, 256)
  oproj_kernel<<<goproj, 32, 0, stream>>>(ctxws, woT, x, pre);
  ln_kernel<<<M_TOT, 256, 0, stream>>>(pre, gamma, beta, out_ptr);
}
